// GraphTransformer_51848845197773
// MI455X (gfx1250) — compile-verified, matrix operand order checked by reading, not run on hardware
//
#include <hip/hip_runtime.h>
#include <hip/hip_bf16.h>

// ---------------- types ----------------
typedef __attribute__((ext_vector_type(8)))  float    v8f;
typedef __attribute__((ext_vector_type(16))) __bf16   v16bf;
typedef __attribute__((ext_vector_type(4)))  unsigned vu4;

union Frag { vu4 u[2]; v16bf v; };
static_assert(sizeof(Frag) == 32, "frag size");

#define NEGF (-1e20f)

// ---------------- TDM support (CDNA5 Tensor Data Mover) ----------------
#if __has_builtin(__builtin_amdgcn_tensor_load_to_lds) && __has_builtin(__builtin_amdgcn_s_wait_tensorcnt)
#define USE_TDM 1
typedef __attribute__((ext_vector_type(4))) unsigned tdm_v4u;
typedef __attribute__((ext_vector_type(8))) int      tdm_v8i;
typedef __attribute__((ext_vector_type(4))) int      tdm_v4i;
typedef __attribute__((address_space(3))) unsigned char lds_byte_t;

__device__ __forceinline__ unsigned lds_addr_of(const void* p) {
  return (unsigned)(unsigned long long)(lds_byte_t*)p;   // addrspacecast -> LDS byte offset
}

// DMA a 64-row x 128-col bf16 tile (row stride = K elements) from global into LDS.
// D# per CDNA5 ISA (2-D tensor, groups 2/3 zero).
// 6-arg builtin variant: (g0 v4u, g1 v8i, g2 v4i, g3 v4i, v8i, cpol)
__device__ __forceinline__ void tdm_load_tile64x128(const unsigned short* gsrc,
                                                    unsigned lds_off, int K) {
  unsigned long long ga = (unsigned long long)(size_t)gsrc;
  tdm_v4u g0;
  g0[0] = 1u;                                        // count=1, user descriptor
  g0[1] = lds_off;                                   // lds_addr (bytes)
  g0[2] = (unsigned)(ga & 0xffffffffu);              // global_addr[31:0]
  g0[3] = (unsigned)((ga >> 32) & 0x01ffffffu)       // global_addr[56:32]
        | 0x80000000u;                               // type = 2 ("image")
  tdm_v8i g1;
  g1[0] = 0x00010000;                                // data_size = 1 (2 bytes), no multicast
  g1[1] = (int)(128u << 16);                         // tensor_dim0 = 128
  g1[2] = (int)(64u  << 16);                         // tensor_dim1 = 64
  g1[3] = (int)(128u << 16);                         // tile_dim0 = 128
  g1[4] = 64;                                        // tile_dim1 = 64 (tile_dim2 = 0)
  g1[5] = K;                                         // tensor_dim0_stride[31:0]
  g1[6] = 0;                                         // stride hi / dim1_stride lo
  g1[7] = 0;
  tdm_v4i gz = {0, 0, 0, 0};                         // dims 2..4 unused
  tdm_v8i gz8 = {0, 0, 0, 0, 0, 0, 0, 0};
  __builtin_amdgcn_tensor_load_to_lds(g0, g1, gz, gz, gz8, 0);
}
#endif

// ---------------- helpers ----------------
__device__ __forceinline__ unsigned short f2bf(float f) {
  union { float f; unsigned u; } c; c.f = f;
  unsigned u = c.u;
  unsigned r = u + 0x7FFFu + ((u >> 16) & 1u);   // round-to-nearest-even
  return (unsigned short)(r >> 16);
}
__device__ __forceinline__ float bf2f(unsigned short h) {
  union { unsigned u; float f; } c; c.u = ((unsigned)h) << 16;
  return c.f;
}
__device__ __forceinline__ float wred_sum(float v) {
  #pragma unroll
  for (int m = 16; m; m >>= 1) v += __shfl_xor(v, m, 32);
  return v;
}
__device__ __forceinline__ float wred_max(float v) {
  #pragma unroll
  for (int m = 16; m; m >>= 1) v = fmaxf(v, __shfl_xor(v, m, 32));
  return v;
}

// ---------------- prep: embedding, pair distances, validity ----------------
__global__ __launch_bounds__(256) void prep_kernel(
    const int* __restrict__ atype, const float* __restrict__ pos,
    const float* __restrict__ emb, float* __restrict__ Z,
    float* __restrict__ D, float* __restrict__ valid)
{
  const int b = blockIdx.x, tid = threadIdx.x;
  __shared__ float px[64], py[64], pz[64];
  __shared__ int ty[64];
  if (tid < 64) {
    int t = atype[b * 64 + tid];
    ty[tid] = t;
    valid[b * 64 + tid] = (t > 0) ? 1.f : 0.f;
    const float* p = pos + ((size_t)b * 64 + tid) * 3;
    px[tid] = p[0]; py[tid] = p[1]; pz[tid] = p[2];
  }
  __syncthreads();
  for (int i = tid; i < 64 * 64; i += 256) {
    int n = i >> 6, m = i & 63;
    float dx = px[n] - px[m], dy = py[n] - py[m], dz = pz[n] - pz[m];
    D[(size_t)b * 4096 + i] = dx * dx + dy * dy + dz * dz;  // == |pn|^2+|pm|^2-2 pn.pm
  }
  for (int i = tid; i < 64 * 640; i += 256) {
    int n = i / 640, d = i - n * 640;
    Z[(size_t)b * 40960 + i] = emb[ty[n] * 640 + d];
  }
}

// ---------------- f32 -> bf16 conversion ----------------
__global__ __launch_bounds__(256) void cvt_kernel(
    const float* __restrict__ src, unsigned short* __restrict__ dst, int n)
{
  for (int i = blockIdx.x * 256 + threadIdx.x; i < n; i += gridDim.x * 256)
    dst[i] = f2bf(src[i]);
}

// ---------------- layernorm (wave per row of 640) ----------------
__global__ __launch_bounds__(256) void ln_kernel(
    const float* __restrict__ Z, const float* __restrict__ gw,
    const float* __restrict__ bw, unsigned short* __restrict__ out)
{
  const int tid = threadIdx.x, wave = tid >> 5, lane = tid & 31;
  const int row = blockIdx.x * 8 + wave;                // B*N rows
  const float* x = Z + (size_t)row * 640;
  float v[20], s = 0.f, s2 = 0.f;
  #pragma unroll
  for (int i = 0; i < 20; ++i) { v[i] = x[i * 32 + lane]; s += v[i]; s2 += v[i] * v[i]; }
  s = wred_sum(s); s2 = wred_sum(s2);
  float mu = s * (1.f / 640.f);
  float var = s2 * (1.f / 640.f) - mu * mu;
  float rs = rsqrtf(var + 1e-5f);
  unsigned short* o = out + (size_t)row * 640;
  #pragma unroll
  for (int i = 0; i < 20; ++i) {
    int c = i * 32 + lane;
    o[c] = f2bf((v[i] - mu) * rs * gw[c] + bw[c]);
  }
}

// ---------------- generic WMMA GEMM ----------------
// C[b] (Mtot x Nout) = epi( A[b] (Mtot x K, bf16 row-major) * W^T (W is Nout x K bf16 row-major)
//                           + bias (+ resid) )
// EPI 0: +bias, store bf16 ; EPI 1: +bias, relu, store bf16 ; EPI 2: (+bias)+resid, store f32
// A K-panels (64x128 bf16) are staged into LDS by the Tensor Data Mover (double-buffered,
// TENSORcnt-tracked); B fragments stream per-lane from L2-resident bf16 weights.
template<int EPI>
__global__ __launch_bounds__(256) void gemm_kernel(
    const unsigned short* __restrict__ A, const unsigned short* __restrict__ W,
    const float* __restrict__ bias, const float* __restrict__ resid,
    void* __restrict__ Out, int Mtot, int K, int Nout)
{
  const int b = blockIdx.x, nblk = blockIdx.y, mblk = blockIdx.z;
  const int tid = threadIdx.x;
  const int wave = tid >> 5, lane = tid & 31;
  const int lh = lane >> 4, l16 = lane & 15;
  const int mt = wave & 3, nh = wave >> 2;              // 4 m-tiles x 2 n-halves

  __shared__ __align__(16) unsigned short Apan[2][64 * 128];   // 2 x 16 KB K-panels

  const unsigned short* Ablk = A + ((size_t)b * Mtot + (size_t)mblk * 64) * K;

  v8f acc0 = {}; v8f acc1 = {};
  const int npan = K >> 7;                              // K multiple of 128

#ifdef USE_TDM
  const bool w0 = (wave == 0);
  if (w0) tdm_load_tile64x128(Ablk, lds_addr_of(&Apan[0][0]), K);
#else
  const int srow = tid >> 2;
  const int scol = (tid & 3) * 32;
#endif

  for (int kp = 0; kp < npan; ++kp) {
#ifdef USE_TDM
    if (w0) __builtin_amdgcn_s_wait_tensorcnt(0);       // panel kp resident in LDS
    __syncthreads();                                    // publish to all waves; WAR for kp+1
    if (w0 && (kp + 1) < npan)
      tdm_load_tile64x128(Ablk + (size_t)(kp + 1) * 128,
                          lds_addr_of(&Apan[(kp + 1) & 1][0]), K);
    const unsigned short* pan = &Apan[kp & 1][0];
#else
    { // synchronous staging fallback
      const unsigned short* src = Ablk + (size_t)srow * K + (size_t)kp * 128 + scol;
      const vu4* s4 = (const vu4*)src;
      vu4* dst = (vu4*)&Apan[kp & 1][srow * 128 + scol];
      dst[0] = s4[0]; dst[1] = s4[1]; dst[2] = s4[2]; dst[3] = s4[3];
      __builtin_prefetch(src + 128, 0, 1);              // next K-panel
    }
    __syncthreads();
    const unsigned short* pan = &Apan[kp & 1][0];
#endif
    #pragma unroll
    for (int ks = 0; ks < 4; ++ks) {
      Frag af, bf0, bf1;
      const int aoff = (mt * 16 + l16) * 128 + ks * 32 + lh * 8;
      af.u[0] = *(const vu4*)&pan[aoff];
      af.u[1] = *(const vu4*)&pan[aoff + 16];
      const size_t kabs = (size_t)kp * 128 + ks * 32 + lh * 8;
      const int n0 = nblk * 64 + (nh * 2) * 16 + l16;
      const unsigned short* w0p = W + (size_t)n0 * K + kabs;
      const unsigned short* w1p = w0p + (size_t)16 * K;
      bf0.u[0] = *(const vu4*)w0p; bf0.u[1] = *(const vu4*)(w0p + 16);
      bf1.u[0] = *(const vu4*)w1p; bf1.u[1] = *(const vu4*)(w1p + 16);
      acc0 = __builtin_amdgcn_wmma_f32_16x16x32_bf16(false, af.v, false, bf0.v,
                                                     (short)0, acc0, false, false);
      acc1 = __builtin_amdgcn_wmma_f32_16x16x32_bf16(false, af.v, false, bf1.v,
                                                     (short)0, acc1, false, false);
    }
#ifndef USE_TDM
    __syncthreads();
#endif
  }

  #pragma unroll
  for (int t2 = 0; t2 < 2; ++t2) {
    v8f acc = t2 ? acc1 : acc0;
    const int col = nblk * 64 + (nh * 2 + t2) * 16 + l16;
    const float bv = bias ? bias[col] : 0.f;
    #pragma unroll
    for (int r = 0; r < 8; ++r) {
      const int row = mblk * 64 + mt * 16 + lh * 8 + r;
      const size_t oidx = ((size_t)b * Mtot + row) * (size_t)Nout + col;
      float v = acc[r] + bv;
      if constexpr (EPI == 0) {
        ((unsigned short*)Out)[oidx] = f2bf(v);
      } else if constexpr (EPI == 1) {
        ((unsigned short*)Out)[oidx] = f2bf(fmaxf(v, 0.f));
      } else {
        ((float*)Out)[oidx] = v + resid[oidx];
      }
    }
  }
}

// ---------------- attention per (batch, head) ----------------
// qkv row layout: [b, n, 1920]; head h occupies cols [h*192, h*192+192) = V | Q | K (64 each)
__global__ __launch_bounds__(256) void attn_kernel(
    const unsigned short* __restrict__ qkv, const float* __restrict__ D,
    const float* __restrict__ valid, const float* __restrict__ gamma,
    unsigned short* __restrict__ attnout)
{
  const int b = blockIdx.x, h = blockIdx.y;
  const int tid = threadIdx.x, wave = tid >> 5, lane = tid & 31;
  const int lh = lane >> 4, l16 = lane & 15;

  __shared__ __align__(16) unsigned short Qs[64 * 64];
  __shared__ __align__(16) unsigned short Ks[64 * 64];
  __shared__ __align__(16) unsigned short Vt[64 * 64];   // V transposed (j, m)
  __shared__ __align__(16) unsigned short Sb[64 * 64];
  __shared__ float Ss[64 * 64];
  __shared__ float vld[64];

  const unsigned short* base = qkv + (size_t)b * 64 * 1920 + h * 192;
  for (int i = tid; i < 4096; i += 256) {
    int n = i >> 6, j = i & 63;
    const unsigned short* rp = base + (size_t)n * 1920;
    Qs[n * 64 + j] = rp[64 + j];     // Z3 = query
    Ks[n * 64 + j] = rp[128 + j];    // Z4 = key
    Vt[j * 64 + n] = rp[j];          // Z2 = value, transposed
  }
  if (tid < 64) vld[tid] = valid[b * 64 + tid];
  __syncthreads();

  const float g = gamma[0];
  const int mt = wave & 3, nb = wave >> 2;

  { // S = Q * K^T   (scores, f32 accum)
    v8f a0 = {}, a1 = {};
    #pragma unroll
    for (int ks = 0; ks < 2; ++ks) {
      Frag af, b0, b1;
      const int ao = (mt * 16 + l16) * 64 + ks * 32 + lh * 8;
      af.u[0] = *(const vu4*)&Qs[ao]; af.u[1] = *(const vu4*)&Qs[ao + 16];
      const int bo0 = ((nb * 2) * 16 + l16) * 64 + ks * 32 + lh * 8;
      const int bo1 = bo0 + 16 * 64;
      b0.u[0] = *(const vu4*)&Ks[bo0]; b0.u[1] = *(const vu4*)&Ks[bo0 + 16];
      b1.u[0] = *(const vu4*)&Ks[bo1]; b1.u[1] = *(const vu4*)&Ks[bo1 + 16];
      a0 = __builtin_amdgcn_wmma_f32_16x16x32_bf16(false, af.v, false, b0.v, (short)0, a0, false, false);
      a1 = __builtin_amdgcn_wmma_f32_16x16x32_bf16(false, af.v, false, b1.v, (short)0, a1, false, false);
    }
    #pragma unroll
    for (int t2 = 0; t2 < 2; ++t2) {
      v8f acc = t2 ? a1 : a0;
      const int col = (nb * 2 + t2) * 16 + l16;
      #pragma unroll
      for (int r = 0; r < 8; ++r) {
        const int row = mt * 16 + lh * 8 + r;
        float s = acc[r] * 0.125f - g * D[((size_t)b * 64 + row) * 64 + col];
        if (vld[row] * vld[col] == 0.f) s = NEGF;
        Ss[row * 64 + col] = s;
      }
    }
  }
  __syncthreads();

  { // row softmax * mask -> Sb (bf16)
    #pragma unroll
    for (int rr = 0; rr < 8; ++rr) {
      const int row = wave * 8 + rr;
      float x0 = Ss[row * 64 + lane], x1 = Ss[row * 64 + 32 + lane];
      float mx = wred_max(fmaxf(x0, x1));
      float e0 = __expf(x0 - mx), e1 = __expf(x1 - mx);
      float sm = wred_sum(e0 + e1);
      float inv = 1.0f / sm;
      float vr = vld[row];
      Sb[row * 64 + lane]      = f2bf(e0 * inv * vr * vld[lane]);
      Sb[row * 64 + 32 + lane] = f2bf(e1 * inv * vr * vld[lane + 32]);
    }
  }
  __syncthreads();

  { // attn = Sb * V ; leaky relu ; store bf16
    v8f a0 = {}, a1 = {};
    #pragma unroll
    for (int ks = 0; ks < 2; ++ks) {
      Frag af, b0, b1;
      const int ao = (mt * 16 + l16) * 64 + ks * 32 + lh * 8;
      af.u[0] = *(const vu4*)&Sb[ao]; af.u[1] = *(const vu4*)&Sb[ao + 16];
      const int bo0 = ((nb * 2) * 16 + l16) * 64 + ks * 32 + lh * 8;
      const int bo1 = bo0 + 16 * 64;
      b0.u[0] = *(const vu4*)&Vt[bo0]; b0.u[1] = *(const vu4*)&Vt[bo0 + 16];
      b1.u[0] = *(const vu4*)&Vt[bo1]; b1.u[1] = *(const vu4*)&Vt[bo1 + 16];
      a0 = __builtin_amdgcn_wmma_f32_16x16x32_bf16(false, af.v, false, b0.v, (short)0, a0, false, false);
      a1 = __builtin_amdgcn_wmma_f32_16x16x32_bf16(false, af.v, false, b1.v, (short)0, a1, false, false);
    }
    #pragma unroll
    for (int t2 = 0; t2 < 2; ++t2) {
      v8f acc = t2 ? a1 : a0;
      const int col = (nb * 2 + t2) * 16 + l16;
      #pragma unroll
      for (int r = 0; r < 8; ++r) {
        const int row = mt * 16 + lh * 8 + r;
        float v = acc[r];
        v = (v > 0.f) ? v : 0.01f * v;
        attnout[((size_t)b * 64 + row) * 640 + h * 64 + col] = f2bf(v);
      }
    }
  }
}

// ---------------- gather pair features ----------------
__global__ __launch_bounds__(256) void gather_kernel(
    const float* __restrict__ Z, const int* __restrict__ scc,
    unsigned short* __restrict__ X)
{
  const int b = blockIdx.x, tid = threadIdx.x;
  for (int i = tid; i < 128 * 1280; i += 256) {
    int p = i / 1280, c = i - p * 1280;
    int which = (c >= 640) ? 1 : 0;
    int n = scc[((size_t)b * 128 + p) * 2 + which];
    int cc = which ? (c - 640) : c;
    X[(size_t)b * 128 * 1280 + i] = f2bf(Z[((size_t)b * 64 + n) * 640 + cc]);
  }
}

// ---------------- final dot: y = H . m2_w + m2_b ----------------
__global__ __launch_bounds__(256) void head_kernel(
    const unsigned short* __restrict__ H, const float* __restrict__ w,
    const float* __restrict__ bsc, float* __restrict__ y)
{
  const int tid = threadIdx.x, wave = tid >> 5, lane = tid & 31;
  const int row = blockIdx.x * 8 + wave;                // B*P rows
  const unsigned short* hr = H + (size_t)row * 1280;
  float s = 0.f;
  #pragma unroll
  for (int i = 0; i < 40; ++i) { int c = i * 32 + lane; s += bf2f(hr[c]) * w[c]; }
  s = wred_sum(s);
  if (lane == 0) y[row] = s + bsc[0];
}

// ---------------- host launch ----------------
extern "C" void kernel_launch(void* const* d_in, const int* in_sizes, int n_in,
                              void* d_out, int out_size, void* d_ws, size_t ws_size,
                              hipStream_t stream) {
  (void)in_sizes; (void)n_in; (void)out_size; (void)ws_size;
  // input order: atom_type, atom_pos, scc_idx, emb, 8 x {qkv_w,qkv_b,o_w,p1_w,p1_b,
  //              p2_w,p2_b,ln1_g,ln1_b,ln2_g,ln2_b,gamma}, m1_w, m1_b, m2_w, m2_b
  const int*   atype = (const int*)  d_in[0];
  const float* apos  = (const float*)d_in[1];
  const int*   scc   = (const int*)  d_in[2];
  const float* emb   = (const float*)d_in[3];
  auto L = [&](int l, int k) -> const float* { return (const float*)d_in[4 + l * 12 + k]; };
  const float* m1w = (const float*)d_in[100];
  const float* m1b = (const float*)d_in[101];
  const float* m2w = (const float*)d_in[102];
  const float* m2b = (const float*)d_in[103];

  // workspace layout
  char* wsb = (char*)d_ws;
  size_t off = 0;
  auto take = [&](size_t bytes) -> char* {
    off = (off + 255) & ~(size_t)255;
    char* p = wsb + off; off += bytes; return p;
  };
  unsigned short *Wq[8], *Wo[8], *Wp1[8], *Wp2[8];
  for (int l = 0; l < 8; ++l) {
    Wq[l]  = (unsigned short*)take((size_t)1920 * 640  * 2);
    Wo[l]  = (unsigned short*)take((size_t)640  * 640  * 2);
    Wp1[l] = (unsigned short*)take((size_t)2560 * 640  * 2);
    Wp2[l] = (unsigned short*)take((size_t)640  * 2560 * 2);
  }
  unsigned short* Wm1  = (unsigned short*)take((size_t)1280 * 1280 * 2);
  float*          Zb   = (float*)         take((size_t)256 * 64 * 640 * 4);
  unsigned short* Znb  = (unsigned short*)take((size_t)256 * 64 * 640 * 2);
  unsigned short* buf1 = (unsigned short*)take((size_t)256 * 128 * 1280 * 2); // qkv / x
  unsigned short* buf2 = (unsigned short*)take((size_t)256 * 128 * 1280 * 2); // attn / h / mh
  float*          Db   = (float*)         take((size_t)256 * 64 * 64 * 4);
  float*          vb   = (float*)         take((size_t)256 * 64 * 4);

  // prep: embedding, distances, mask; weight conversion f32->bf16 (weights then live in L2)
  prep_kernel<<<256, 256, 0, stream>>>(atype, apos, emb, Zb, Db, vb);
  for (int l = 0; l < 8; ++l) {
    cvt_kernel<<<2048, 256, 0, stream>>>(L(l, 0), Wq[l],  1920 * 640);
    cvt_kernel<<<2048, 256, 0, stream>>>(L(l, 2), Wo[l],  640  * 640);
    cvt_kernel<<<2048, 256, 0, stream>>>(L(l, 3), Wp1[l], 2560 * 640);
    cvt_kernel<<<2048, 256, 0, stream>>>(L(l, 5), Wp2[l], 640  * 2560);
  }
  cvt_kernel<<<2048, 256, 0, stream>>>(m1w, Wm1, 1280 * 1280);

  // 8 transformer layers
  for (int l = 0; l < 8; ++l) {
    ln_kernel<<<2048, 256, 0, stream>>>(Zb, L(l, 7), L(l, 8), Znb);
    gemm_kernel<0><<<dim3(256, 30, 1), 256, 0, stream>>>(Znb, Wq[l], L(l, 1), nullptr,
                                                         buf1, 64, 640, 1920);
    attn_kernel<<<dim3(256, 10), 256, 0, stream>>>(buf1, Db, vb, L(l, 11), buf2);
    gemm_kernel<2><<<dim3(256, 10, 1), 256, 0, stream>>>(buf2, Wo[l], nullptr, Zb,
                                                         Zb, 64, 640, 640);
    ln_kernel<<<2048, 256, 0, stream>>>(Zb, L(l, 9), L(l, 10), Znb);
    gemm_kernel<1><<<dim3(256, 40, 1), 256, 0, stream>>>(Znb, Wp1[l], L(l, 4), nullptr,
                                                         buf2, 64, 640, 2560);
    gemm_kernel<2><<<dim3(256, 10, 1), 256, 0, stream>>>(buf2, Wp2[l], L(l, 6), Zb,
                                                         Zb, 64, 2560, 640);
  }

  // readout head
  gather_kernel<<<256, 256, 0, stream>>>(Zb, scc, buf1);
  gemm_kernel<1><<<dim3(256, 20, 2), 256, 0, stream>>>(buf1, Wm1, m1b, nullptr,
                                                       buf2, 128, 1280, 1280);
  head_kernel<<<4096, 256, 0, stream>>>(buf2, m2w, m2b, (float*)d_out);
}
